// PureSelfAttention_1614907703859
// MI455X (gfx1250) — matrix operand order repaired and checked by reading, not compiled, not run on hardware
//
#include <hip/hip_runtime.h>
#include <hip/hip_bf16.h>
#include <math.h>

// ---------------- problem constants ----------------
#define BQ 4
#define TT 2048
#define DD 1024
#define HH 16
#define DHD 64
#define MM (BQ * TT)   // 8192 rows

typedef __attribute__((ext_vector_type(16))) _Float16 v16h;
typedef __attribute__((ext_vector_type(8)))  _Float16 v8h;
typedef __attribute__((ext_vector_type(4)))  _Float16 v4h;
typedef __attribute__((ext_vector_type(8)))  float    v8f;
typedef __attribute__((ext_vector_type(4)))  unsigned int v4u;
typedef __attribute__((ext_vector_type(8)))  unsigned int v8u;

// A-matrix 16x32 f16 layout: lane l -> row (l&15); K chunks [kb..kb+7] and
// [kb+16..kb+23], kb = 8*(l>=16). Two contiguous 16B loads.
static __device__ __forceinline__ v16h ld_a16(const _Float16* p) {
    v8h lo = *(const v8h*)(p);
    v8h hi = *(const v8h*)(p + 16);
    return __builtin_shufflevector(lo, hi, 0, 1, 2, 3, 4, 5, 6, 7,
                                           8, 9, 10, 11, 12, 13, 14, 15);
}
// B-matrix 32x16 f16 layout: lane l -> col (l&15); K chunks [kb..kb+7] and
// [kb+16..kb+23], kb = 8*(l>=16). Two 16B loads (LDS rows are 16B- but not 32B-aligned).
static __device__ __forceinline__ v16h ld_b16(const _Float16* p) {
    v8h lo = *(const v8h*)(p);
    v8h hi = *(const v8h*)(p + 16);
    return __builtin_shufflevector(lo, hi, 0, 1, 2, 3, 4, 5, 6, 7,
                                           8, 9, 10, 11, 12, 13, 14, 15);
}
static __device__ __forceinline__ v8f wmma_f16(v16h a, v16h b, v8f c) {
    return __builtin_amdgcn_wmma_f32_16x16x32_f16(
        false, a, false, b, (short)0, c, false, false);
}

// ---------------- Tensor Data Mover: 2D tile -> LDS ----------------
// Builds the D# per CDNA5 ISA ch.8 and issues tensor_load_to_lds with the
// 2-SGPR-group form (VADDR2/VADDR3 = NULL -> tensors up to 2D).
// data_size = 2 bytes. pad_enable inserts pad_amt(code) DWORDs of LDS padding
// every 2^(pi_code+1) DWORDs stored (bank-conflict avoidance).
static __device__ __forceinline__ void tdm_load_2d(
    const _Float16* gsrc, const _Float16* lds_dst,
    unsigned dim0, unsigned dim1, unsigned stride0,
    unsigned tile0, unsigned tile1,
    unsigned pi_code, unsigned pa_code) {
    unsigned long long ga = (unsigned long long)(uintptr_t)gsrc;
    unsigned ldsoff = (unsigned)(uintptr_t)lds_dst;  // generic LDS ptr low 32 = LDS offset
    v4u g0;
    g0[0] = 1u;                                        // count=1, no gather
    g0[1] = ldsoff;                                    // lds_addr (bytes)
    g0[2] = (unsigned)ga;                              // global_addr[31:0]
    g0[3] = (unsigned)((ga >> 32) & 0x01FFFFFFu) | (2u << 30);  // addr[56:32] | type=2
    v8u g1;
    g1[0] = (1u << 16) | (1u << 20) | (pi_code << 22) | (pa_code << 25);
    g1[1] = (dim0 & 0xFFFFu) << 16;                    // tensor_dim0[15:0]
    g1[2] = (dim0 >> 16) | ((dim1 & 0xFFFFu) << 16);   // dim0 hi | dim1 lo
    g1[3] = (dim1 >> 16) | (tile0 << 16);              // dim1 hi | tile_dim0
    g1[4] = tile1;                                     // tile_dim1 (tile_dim2=0)
    g1[5] = stride0;                                   // tensor_dim0_stride[31:0]
    g1[6] = 0u;                                        // stride0 hi | stride1 lo
    g1[7] = 0u;
    asm volatile("tensor_load_to_lds %0, %1" :: "s"(g0), "s"(g1) : "memory");
}

// ---------------- fp32 -> f16 convert (x) ----------------
__global__ void k_cvt(const float* __restrict__ in, _Float16* __restrict__ out, int n) {
    int i = (blockIdx.x * blockDim.x + threadIdx.x) * 4;
    if (i + 3 < n) {
        float4 v = *(const float4*)(in + i);
        v4h h = { (_Float16)v.x, (_Float16)v.y, (_Float16)v.z, (_Float16)v.w };
        *(v4h*)(out + i) = h;
    }
}

// ---------------- fp32 -> f16 + transpose (weights) ----------------
__global__ void k_wcvt_t(const float* __restrict__ w, _Float16* __restrict__ wt) {
    int n = blockIdx.x * 32 + threadIdx.x;
    int k0 = blockIdx.y * 32 + threadIdx.y;
    #pragma unroll
    for (int i = 0; i < 32; i += 8) {
        int k = k0 + i;
        wt[(size_t)n * DD + k] = (_Float16)w[(size_t)k * DD + n];
    }
}

// ---------------- WMMA GEMM with TDM double-buffered LDS staging ----------
// C[M,N] = A[M,K] * W[K,N]; A f16 row-major, Wt = W^T f16 [N][K].
// Block = 4 waves; block tile 128x128; K-step 32. LDS rows padded to 40 halves.
// MODE 0: f16 out. MODE 1: f16 out transposed to [B,H,Dh,T] (V). MODE 2: f32.
#define GSTRIDE 40   // 32 halves + 8 pad halves (64B rows + 16B pad)
template <int MODE>
__global__ void __launch_bounds__(128)
k_gemm(const _Float16* __restrict__ A, const _Float16* __restrict__ Wt,
       void* __restrict__ outv) {
    __shared__ _Float16 As[2][128 * GSTRIDE];
    __shared__ _Float16 Bs[2][128 * GSTRIDE];

    const int lane = threadIdx.x & 31;
    const int wv   = threadIdx.x >> 5;
    const int ln   = lane & 15;
    const int hi   = lane >> 4;
    const int rowblk = blockIdx.y * 128;
    const int colblk = blockIdx.x * 128;
    const int warow = (wv >> 1) * 64;
    const int wacol = (wv & 1) * 64;
    const int kbA = hi * 8;
    const int kbB = hi * 8;

    if (wv == 0) {   // prologue DMA: buffer 0, kk=0 (rows x 32-half slabs)
        tdm_load_2d(A  + (size_t)rowblk * DD, As[0], DD, MM - rowblk, DD, 32, 128, 3, 3);
        tdm_load_2d(Wt + (size_t)colblk * DD, Bs[0], DD, DD - colblk, DD, 32, 128, 3, 3);
    }

    v8f acc[4][4] = {};

    for (int kk = 0; kk < DD; kk += 32) {
        const int buf = (kk >> 5) & 1;
        if (wv == 0) __builtin_amdgcn_s_wait_tensorcnt(0);
        __syncthreads();                       // staged tiles ready for all waves
        if (wv == 0 && kk + 32 < DD) {         // prefetch next K-slab into other buffer
            tdm_load_2d(A  + (size_t)rowblk * DD + kk + 32, As[buf ^ 1],
                        DD - kk - 32, MM - rowblk, DD, 32, 128, 3, 3);
            tdm_load_2d(Wt + (size_t)colblk * DD + kk + 32, Bs[buf ^ 1],
                        DD - kk - 32, DD - colblk, DD, 32, 128, 3, 3);
        }
        const _Float16* Ab = As[buf];
        const _Float16* Bb = Bs[buf];
        v16h at[4], bt[4];
        #pragma unroll
        for (int a = 0; a < 4; ++a)
            at[a] = ld_a16(Ab + (warow + 16 * a + ln) * GSTRIDE + kbA);
        #pragma unroll
        for (int c = 0; c < 4; ++c)
            bt[c] = ld_b16(Bb + (wacol + 16 * c + ln) * GSTRIDE + kbB);
        #pragma unroll
        for (int a = 0; a < 4; ++a)
            #pragma unroll
            for (int c = 0; c < 4; ++c)
                acc[a][c] = wmma_f16(at[a], bt[c], acc[a][c]);
    }

    #pragma unroll
    for (int a = 0; a < 4; ++a) {
        #pragma unroll
        for (int c = 0; c < 4; ++c) {
            #pragma unroll
            for (int r = 0; r < 8; ++r) {
                int m = rowblk + warow + 16 * a + hi * 8 + r;
                int n = colblk + wacol + 16 * c + ln;
                float val = acc[a][c][r];
                if (MODE == 0) {
                    ((_Float16*)outv)[(size_t)m * DD + n] = (_Float16)val;
                } else if (MODE == 1) {
                    int b = m >> 11, t = m & (TT - 1);
                    int h = n >> 6,  d = n & (DHD - 1);
                    ((_Float16*)outv)[(((size_t)b * HH + h) * DHD + d) * TT + t] =
                        (_Float16)val;
                } else {
                    ((float*)outv)[(size_t)m * DD + n] = val;
                }
            }
        }
    }
}

// ---------------- flash attention with TDM-staged K/V ----------------
// Q,K: f16 [B,T,D].  Vt: f16 [B,H,Dh,T].  Y: f16 [B,T,D].
// 4 waves/block share one (b,h); each owns a 16-query tile; K/V 32-key blocks
// are DMA'd to LDS once per block and shared by all 4 waves.
#define KSTRIDE 72   // 64 halves + 8 pad (128B rows + 16B pad)
#define VSTRIDE 40   // 32 halves + 8 pad (64B rows + 16B pad)
__global__ void __launch_bounds__(128)
k_attn(const _Float16* __restrict__ Q, const _Float16* __restrict__ K,
       const _Float16* __restrict__ Vt, _Float16* __restrict__ Y) {
    __shared__ _Float16 Ks[2][32 * KSTRIDE];   // [key][d]
    __shared__ _Float16 Vs[2][64 * VSTRIDE];   // [d][key]
    __shared__ _Float16 plds[4][16 * 32];      // per-wave P staging

    const int lane = threadIdx.x & 31;
    const int wv   = threadIdx.x >> 5;
    const int ln   = lane & 15;
    const int hi   = lane >> 4;

    const int tile = blockIdx.x * 4 + wv;      // 8192 tiles; 4 per block share bh
    const int bh   = tile >> 7;
    const int q0   = (tile & 127) * 16;
    const int b    = bh >> 4;
    const int h    = bh & 15;

    const _Float16* Kbase = K  + ((size_t)b * TT) * DD + h * DHD;        // [key][d], stride DD
    const _Float16* Vbase = Vt + ((size_t)b * HH + h) * DHD * TT;        // [d][key], stride TT

    if (wv == 0) {   // prologue: stage first 32-key block
        tdm_load_2d(Kbase, Ks[0], DD, TT, DD, 64, 32, 4, 3);
        tdm_load_2d(Vbase, Vs[0], TT, DHD, TT, 32, 64, 3, 3);
    }

    // Q A-tiles resident in registers (Dh=64 -> 2 tiles of K=32).
    const size_t qrow = ((size_t)b * TT + q0 + ln) * DD + h * DHD;
    v16h qa[2];
    #pragma unroll
    for (int d2 = 0; d2 < 2; ++d2)
        qa[d2] = ld_a16(Q + qrow + 32 * d2 + hi * 8);

    v8f oacc[4] = {};
    float mrow[8], lrow[8];
    #pragma unroll
    for (int r = 0; r < 8; ++r) { mrow[r] = -1e30f; lrow[r] = 0.0f; }

    _Float16* P = plds[wv];
    const float scale = 0.125f;   // 1/sqrt(64)

    for (int kb = 0; kb < TT; kb += 32) {
        const int buf = (kb >> 5) & 1;
        if (wv == 0) __builtin_amdgcn_s_wait_tensorcnt(0);
        __syncthreads();                            // K/V block ready (+WAR on P)
        if (wv == 0 && kb + 32 < TT) {
            tdm_load_2d(Kbase + (size_t)(kb + 32) * DD, Ks[buf ^ 1],
                        DD, TT - kb - 32, DD, 64, 32, 4, 3);
            tdm_load_2d(Vbase + kb + 32, Vs[buf ^ 1],
                        TT - kb - 32, DHD, TT, 32, 64, 3, 3);
        }
        const _Float16* Kb = Ks[buf];
        const _Float16* Vb = Vs[buf];

        // S = Q * K^T for 32 keys
        v8f s[2] = {};
        #pragma unroll
        for (int c = 0; c < 2; ++c)
            #pragma unroll
            for (int d2 = 0; d2 < 2; ++d2)
                s[c] = wmma_f16(qa[d2],
                                ld_b16(Kb + (16 * c + ln) * KSTRIDE + 32 * d2 + hi * 8),
                                s[c]);

        // online softmax; lanes 0-15 own rows 0-7, lanes 16-31 own rows 8-15
        #pragma unroll
        for (int r = 0; r < 8; ++r) {
            float x0 = s[0][r] * scale;
            float x1 = s[1][r] * scale;
            float mx = fmaxf(x0, x1);
            #pragma unroll
            for (int m = 8; m >= 1; m >>= 1) mx = fmaxf(mx, __shfl_xor(mx, m, 32));
            float mn   = fmaxf(mrow[r], mx);
            float corr = __expf(mrow[r] - mn);
            mrow[r] = mn;
            float p0 = __expf(x0 - mn);
            float p1 = __expf(x1 - mn);
            float rs = p0 + p1;
            #pragma unroll
            for (int m = 8; m >= 1; m >>= 1) rs += __shfl_xor(rs, m, 32);
            lrow[r] = lrow[r] * corr + rs;
            #pragma unroll
            for (int t = 0; t < 4; ++t) oacc[t][r] *= corr;
            int row = hi * 8 + r;
            P[row * 32 + ln]      = (_Float16)p0;
            P[row * 32 + 16 + ln] = (_Float16)p1;
        }
        __syncthreads();                            // P stores visible
        v16h pa = ld_a16(P + ln * 32 + hi * 8);     // P in A layout

        // O += P * V
        #pragma unroll
        for (int t = 0; t < 4; ++t)
            oacc[t] = wmma_f16(pa,
                               ld_b16(Vb + (16 * t + ln) * VSTRIDE + hi * 8),
                               oacc[t]);
    }

    #pragma unroll
    for (int r = 0; r < 8; ++r) {
        float inv = 1.0f / lrow[r];
        int gq = q0 + hi * 8 + r;
        #pragma unroll
        for (int t = 0; t < 4; ++t)
            Y[((size_t)b * TT + gq) * DD + h * DHD + 16 * t + ln] =
                (_Float16)(oacc[t][r] * inv);
    }
}

// ---------------- host launch ----------------
extern "C" void kernel_launch(void* const* d_in, const int* in_sizes, int n_in,
                              void* d_out, int out_size, void* d_ws, size_t ws_size,
                              hipStream_t stream) {
    (void)in_sizes; (void)n_in; (void)out_size; (void)ws_size;
    const float* x  = (const float*)d_in[0];
    const float* Wq = (const float*)d_in[1];
    const float* Wk = (const float*)d_in[2];
    const float* Wv = (const float*)d_in[3];
    const float* Wo = (const float*)d_in[4];

    char* ws = (char*)d_ws;
    const size_t SZ_X = (size_t)MM * DD * sizeof(_Float16);  // 16 MB
    const size_t SZ_W = (size_t)DD * DD * sizeof(_Float16);  //  2 MB
    _Float16* xh  = (_Float16*)(ws);
    _Float16* qh  = (_Float16*)(ws + 1 * SZ_X);
    _Float16* kh  = (_Float16*)(ws + 2 * SZ_X);
    _Float16* vt  = (_Float16*)(ws + 3 * SZ_X);              // [B,H,Dh,T]
    _Float16* yh  = (_Float16*)(ws + 4 * SZ_X);
    _Float16* wqt = (_Float16*)(ws + 5 * SZ_X);
    _Float16* wkt = (_Float16*)(ws + 5 * SZ_X + 1 * SZ_W);
    _Float16* wvt = (_Float16*)(ws + 5 * SZ_X + 2 * SZ_W);
    _Float16* wot = (_Float16*)(ws + 5 * SZ_X + 3 * SZ_W);

    const int nx = MM * DD;
    k_cvt<<<nx / (256 * 4), 256, 0, stream>>>(x, xh, nx);

    dim3 tb(32, 8), tg(DD / 32, DD / 32);
    k_wcvt_t<<<tg, tb, 0, stream>>>(Wq, wqt);
    k_wcvt_t<<<tg, tb, 0, stream>>>(Wk, wkt);
    k_wcvt_t<<<tg, tb, 0, stream>>>(Wv, wvt);
    k_wcvt_t<<<tg, tb, 0, stream>>>(Wo, wot);

    dim3 gg(DD / 128, MM / 128);
    k_gemm<0><<<gg, 128, 0, stream>>>(xh, wqt, qh);
    k_gemm<0><<<gg, 128, 0, stream>>>(xh, wkt, kh);
    k_gemm<1><<<gg, 128, 0, stream>>>(xh, wvt, vt);

    k_attn<<<(BQ * HH * (TT / 16)) / 4, 128, 0, stream>>>(qh, kh, vt, yh);

    k_gemm<2><<<gg, 128, 0, stream>>>(yh, wot, d_out);
}
